// SeqAttention_78091095376383
// MI455X (gfx1250) — matrix operand order repaired and checked by reading, not compiled, not run on hardware
//
#include <hip/hip_runtime.h>

// ---------------------------------------------------------------------------
// out[b] = (Q Kt) V  ==  Q (Kt V)   (softmax/attn_pos are dead code in ref)
//   Stage 1: Tt[b] (1024x1024 bf16) = V[b]^T K[b]   (i.e. (Kt V) transposed)
//   Stage 2: out[b] (2048x1024 f32) = Q[b] * Tt[b]^T
// Transposed intermediate -> stage 2's B operand is a contiguous bf16 copy
// via GLOBAL_LOAD_ASYNC_TO_LDS_B128 (double-buffered, s_wait_asynccnt).
// bf16 WMMA (v_wmma_f32_16x16x32_bf16), fp32 accumulate.
// ---------------------------------------------------------------------------

typedef __attribute__((ext_vector_type(16))) __bf16 v16bf;
typedef __attribute__((ext_vector_type(8)))  __bf16 v8bf;
typedef __attribute__((ext_vector_type(2)))  __bf16 v2bf;
typedef __attribute__((ext_vector_type(8)))  float  v8f;
typedef __attribute__((ext_vector_type(4)))  int    v4i;

union FragU { v16bf v; v8bf h[2]; };

static constexpr int Bb = 8;
static constexpr int Mm = 2048;
static constexpr int Ll = 2048;
static constexpr int Hh = 1024;

static constexpr int BM  = 128;   // C-tile rows per workgroup
static constexpr int BN  = 128;   // C-tile cols per workgroup
static constexpr int BK  = 32;    // k-slab per LDS stage
static constexpr int LDA = 40;    // padded LDS row (bf16); 80B rows keep 16B align

// ---- CDNA5 async global->LDS path (guarded; plain-copy fallback) ----------
#if defined(__has_builtin)
# if __has_builtin(__builtin_amdgcn_global_load_async_to_lds_b128)
#  define HAVE_ASYNC_LDS 1
# endif
#endif
#ifndef HAVE_ASYNC_LDS
# define HAVE_ASYNC_LDS 0
#endif

#if HAVE_ASYNC_LDS
# if defined(__has_builtin) && __has_builtin(__builtin_amdgcn_s_wait_asynccnt)
#  define WAIT_ASYNC(N) __builtin_amdgcn_s_wait_asynccnt(N)
# else
#  define WAIT_ASYNC(N) asm volatile("s_wait_asynccnt " #N ::: "memory")
# endif
#else
# define WAIT_ASYNC(N) ((void)0)
#endif

__device__ __forceinline__ void copy_b128_to_lds(const unsigned short* g,
                                                 unsigned short* l) {
#if HAVE_ASYNC_LDS
  // Probe-learned signature: (v4i* global_src, v4i* lds_dst, int off, int cpol)
  __builtin_amdgcn_global_load_async_to_lds_b128((v4i*)g, (v4i*)l, 0, 0);
#else
  *(uint4*)l = *(const uint4*)g;
#endif
}

// fp32 -> bf16, round-half-up: one VALU add, store from high half.
__device__ __forceinline__ unsigned short f2bf(float x) {
  unsigned int u = __builtin_bit_cast(unsigned int, x) + 0x8000u;
  return (unsigned short)(u >> 16);
}

// pack two fp32 -> packed bf16 pair (hardware packed cvt when available)
__device__ __forceinline__ unsigned int pack2bf(float a, float b) {
#if defined(__has_builtin)
# if __has_builtin(__builtin_amdgcn_cvt_pk_bf16_f32)
  v2bf p = __builtin_amdgcn_cvt_pk_bf16_f32(a, b);
  return __builtin_bit_cast(unsigned int, p);
# else
  return (unsigned int)f2bf(a) | ((unsigned int)f2bf(b) << 16);
# endif
#else
  return (unsigned int)f2bf(a) | ((unsigned int)f2bf(b) << 16);
#endif
}

// Shared inner compute: 8 waves, wave (wm,wn) owns 64x32 of the 128x128 tile.
// LDS A: [row 0..127][k 0..31] ; LDS B: [col 0..127][k 0..31].
__device__ __forceinline__ void mma_tiles(const unsigned short (&lds_a)[BM][LDA],
                                          const unsigned short (&lds_b)[BN][LDA],
                                          int lane, int wm, int wn,
                                          v8f (&acc)[4][2]) {
  const int m = lane & 15;
  const int g = lane >> 4;          // k-half select per ISA fragment layout
  FragU af[4], bf[2];
#pragma unroll
  for (int im = 0; im < 4; ++im) {
    const int row = wm * 64 + im * 16 + m;
    af[im].h[0] = *(const v8bf*)&lds_a[row][8 * g];        // K = 8g..8g+7
    af[im].h[1] = *(const v8bf*)&lds_a[row][16 + 8 * g];   // K = 16+8g..
  }
#pragma unroll
  for (int jn = 0; jn < 2; ++jn) {
    const int col = wn * 32 + jn * 16 + m;
    bf[jn].h[0] = *(const v8bf*)&lds_b[col][8 * g];        // K = 8g..8g+7
    bf[jn].h[1] = *(const v8bf*)&lds_b[col][16 + 8 * g];   // K = 16+8g..
  }
#pragma unroll
  for (int im = 0; im < 4; ++im)
#pragma unroll
    for (int jn = 0; jn < 2; ++jn)
      acc[im][jn] = __builtin_amdgcn_wmma_f32_16x16x32_bf16(
          false, af[im].v, false, bf[jn].v, (short)0, acc[im][jn], false, false);
}

// ---------------- Stage 1: C[i][n] = sum_l A[l][i] * B[l][n]  (bf16 out) ----
// Called as (A=V, B=K): C = V^T K = Tt.  Register-pipelined fp32 loads.
__global__ __launch_bounds__(256) void xtx_kernel(const float* __restrict__ Ap,
                                                  const float* __restrict__ Bp,
                                                  unsigned short* __restrict__ Cp) {
  const int bz = blockIdx.z;
  const int i0 = blockIdx.y * BM;
  const int n0 = blockIdx.x * BN;
  const float* Ab = Ap + (size_t)bz * Ll * Hh;
  const float* Bbp = Bp + (size_t)bz * Ll * Hh;
  unsigned short* Cb = Cp + (size_t)bz * Hh * Hh;

  __shared__ unsigned short lds_a[BM][LDA];
  __shared__ unsigned short lds_b[BN][LDA];

  const int t = threadIdx.x;
  const int lane = t & 31, w = t >> 5;
  const int wm = w >> 2, wn = w & 3;

  v8f acc[4][2];
#pragma unroll
  for (int i = 0; i < 4; ++i)
#pragma unroll
    for (int j = 0; j < 2; ++j)
#pragma unroll
      for (int e = 0; e < 8; ++e) acc[i][j][e] = 0.0f;

  float4 pa[4], pb[4];
  auto load_slab = [&](int k0) {
#pragma unroll
    for (int it = 0; it < 4; ++it) {
      const int f = t + 256 * it, l = f >> 5, c4 = f & 31;
      pa[it] = *(const float4*)(Ab + (size_t)(k0 + l) * Hh + i0 + 4 * c4);
      pb[it] = *(const float4*)(Bbp + (size_t)(k0 + l) * Hh + n0 + 4 * c4);
    }
  };
  auto store_slab = [&]() {
#pragma unroll
    for (int it = 0; it < 4; ++it) {
      const int f = t + 256 * it, l = f >> 5, c4 = f & 31;
      lds_a[4 * c4 + 0][l] = f2bf(pa[it].x);
      lds_a[4 * c4 + 1][l] = f2bf(pa[it].y);
      lds_a[4 * c4 + 2][l] = f2bf(pa[it].z);
      lds_a[4 * c4 + 3][l] = f2bf(pa[it].w);
      lds_b[4 * c4 + 0][l] = f2bf(pb[it].x);
      lds_b[4 * c4 + 1][l] = f2bf(pb[it].y);
      lds_b[4 * c4 + 2][l] = f2bf(pb[it].z);
      lds_b[4 * c4 + 3][l] = f2bf(pb[it].w);
    }
  };

  load_slab(0);
  for (int k0 = 0; k0 < Ll; k0 += BK) {
    store_slab();
    if (k0 + BK < Ll) load_slab(k0 + BK);  // overlap with WMMA below
    __syncthreads();
    mma_tiles(lds_a, lds_b, lane, wm, wn, acc);
    __syncthreads();
  }

  // D layout: VGPR v -> row = v + 8*(lane/16), col = lane%16 (coalesced cols)
  const int nl = lane & 15, gh = lane >> 4;
#pragma unroll
  for (int im = 0; im < 4; ++im)
#pragma unroll
    for (int jn = 0; jn < 2; ++jn)
#pragma unroll
      for (int v = 0; v < 8; ++v) {
        const int row = i0 + wm * 64 + im * 16 + v + 8 * gh;
        const int col = n0 + wn * 32 + jn * 16 + nl;
        Cb[(size_t)row * Hh + col] = f2bf(acc[im][jn][v]);
      }
}

// ---------------- Stage 2: out = Q * Tt^T  (f32 out) -----------------------
// B operand (Tt rows) is contiguous bf16 -> async copy, double-buffered LDS.
__global__ __launch_bounds__(256) void qt_kernel(const float* __restrict__ Qp,
                                                 const unsigned short* __restrict__ Tt,
                                                 float* __restrict__ Op) {
  const int bz = blockIdx.z;
  const int m0 = blockIdx.y * BM;
  const int n0 = blockIdx.x * BN;
  const float* Qb = Qp + (size_t)bz * Mm * Hh;
  const unsigned short* Tb = Tt + (size_t)bz * Hh * Hh;  // Tt[n][k], ld = Hh
  float* Ob = Op + (size_t)bz * Mm * Hh;

  __shared__ unsigned short lds_a[BM][LDA];
  __shared__ unsigned short lds_b[2][BN][LDA];           // double buffered

  const int t = threadIdx.x;
  const int lane = t & 31, w = t >> 5;
  const int wm = w >> 2, wn = w & 3;

  v8f acc[4][2];
#pragma unroll
  for (int i = 0; i < 4; ++i)
#pragma unroll
    for (int j = 0; j < 2; ++j)
#pragma unroll
      for (int e = 0; e < 8; ++e) acc[i][j][e] = 0.0f;

  float4 pa[4];
  auto load_a = [&](int k0) {
#pragma unroll
    for (int it = 0; it < 4; ++it) {
      const int f = t + 256 * it, row = f >> 3, c4 = f & 7;
      pa[it] = *(const float4*)(Qb + (size_t)(m0 + row) * Hh + k0 + 4 * c4);
    }
  };
  auto store_a = [&]() {
#pragma unroll
    for (int it = 0; it < 4; ++it) {
      const int f = t + 256 * it, row = f >> 3, c4 = f & 7;
      uint2 p;
      p.x = pack2bf(pa[it].x, pa[it].y);
      p.y = pack2bf(pa[it].z, pa[it].w);
      *(uint2*)&lds_a[row][4 * c4] = p;
    }
  };
  // Tt slab: 128 n-rows x 32 k = 2 x b128 per thread (16B chunks).
  auto stage_b = [&](int k0, int buf) {
#pragma unroll
    for (int it = 0; it < 2; ++it) {
      const int c = t + 256 * it;       // 0..511 chunks
      const int n = c >> 2, kc = c & 3; // 4 chunks per row
      copy_b128_to_lds(Tb + (size_t)(n0 + n) * Hh + k0 + 8 * kc,
                       &lds_b[buf][n][8 * kc]);
    }
  };

  stage_b(0, 0);
  load_a(0);
  int buf = 0;
  for (int k0 = 0; k0 < Hh; k0 += BK) {
    store_a();
    if (k0 + BK < Hh) {
      load_a(k0 + BK);
      stage_b(k0 + BK, buf ^ 1);
      WAIT_ASYNC(2);   // current slab's 2 transfers done; next 2 in flight
    } else {
      WAIT_ASYNC(0);
    }
    __syncthreads();
    mma_tiles(lds_a, lds_b[buf], lane, wm, wn, acc);
    __syncthreads();
    buf ^= 1;
  }

  const int nl = lane & 15, gh = lane >> 4;
#pragma unroll
  for (int im = 0; im < 4; ++im)
#pragma unroll
    for (int jn = 0; jn < 2; ++jn)
#pragma unroll
      for (int v = 0; v < 8; ++v) {
        const int row = m0 + wm * 64 + im * 16 + v + 8 * gh;
        const int col = n0 + wn * 32 + jn * 16 + nl;
        Ob[(size_t)row * Hh + col] = acc[im][jn][v];
      }
}

extern "C" void kernel_launch(void* const* d_in, const int* in_sizes, int n_in,
                              void* d_out, int out_size, void* d_ws, size_t ws_size,
                              hipStream_t stream) {
  (void)in_sizes; (void)n_in; (void)out_size; (void)ws_size;
  const float* Q = (const float*)d_in[0];    // [B, M, H]
  const float* K = (const float*)d_in[1];    // [B, L, H]
  const float* V = (const float*)d_in[2];    // [B, L, H]
  // d_in[3] = key_pe, d_in[4] = hidden_size: dead w.r.t. the output.
  float* out = (float*)d_out;                // [B, M, H]
  unsigned short* Tt = (unsigned short*)d_ws; // bf16 [B, H, H] transposed, 16 MiB

  // Tt = V^T K  (== (K^T V)^T), then out = Q * Tt^T
  xtx_kernel<<<dim3(Hh / BN, Hh / BM, Bb), 256, 0, stream>>>(V, K, Tt);
  qt_kernel <<<dim3(Hh / BN, Mm / BM, Bb), 256, 0, stream>>>(Q, Tt, out);
}
